// SimpleHeteroGNN_3564822856030
// MI455X (gfx1250) — matrix operand order repaired and checked by reading, not compiled, not run on hardware
//
#include <hip/hip_runtime.h>

#define NU 100000
#define NI 200000
#define EE 500000
#define DIN 128
#define DH  128
#define DOUT 64

typedef __attribute__((ext_vector_type(16))) __bf16          v16bf;
typedef __attribute__((ext_vector_type(16))) unsigned short  v16u;
typedef __attribute__((ext_vector_type(8)))  unsigned int    v8u;
typedef __attribute__((ext_vector_type(8)))  float           v8f;

__device__ __forceinline__ unsigned short f2b(float f) {
  unsigned u = __float_as_uint(f);
  unsigned r = u + 0x7fffu + ((u >> 16) & 1u);   // round-to-nearest-even
  return (unsigned short)(r >> 16);
}
__device__ __forceinline__ float b2f(unsigned short h) {
  return __uint_as_float(((unsigned)h) << 16);
}
__device__ __forceinline__ void atomAddF(float* p, float v) {
  __hip_atomic_fetch_add(p, v, __ATOMIC_RELAXED, __HIP_MEMORY_SCOPE_AGENT);
}

// ---------------- utility kernels ----------------

__global__ void fill_f32(float* __restrict__ p, size_t n, float v) {
  for (size_t i = blockIdx.x * (size_t)blockDim.x + threadIdx.x; i < n;
       i += (size_t)gridDim.x * blockDim.x)
    p[i] = v;
}

__global__ void degree_kernel(const int* __restrict__ idx, float* __restrict__ deg, int n) {
  for (int i = blockIdx.x * blockDim.x + threadIdx.x; i < n; i += gridDim.x * blockDim.x)
    atomAddF(&deg[idx[i]], 1.0f);
}

__global__ void rsqrt_kernel(float* __restrict__ p, size_t n) {
  for (size_t i = blockIdx.x * (size_t)blockDim.x + threadIdx.x; i < n;
       i += (size_t)gridDim.x * blockDim.x)
    p[i] = rsqrtf(fmaxf(p[i], 1.0f));
}

// 4-wide f32 -> packed bf16 (optional ReLU). n4 = n/4. One b128 load + b64 store per thread-iter.
template <bool RELU>
__global__ void cvt_bf16x4(const float* __restrict__ in, unsigned short* __restrict__ out, size_t n4) {
  for (size_t i = blockIdx.x * (size_t)blockDim.x + threadIdx.x; i < n4;
       i += (size_t)gridDim.x * blockDim.x) {
    float4 f = ((const float4*)in)[i];
    if constexpr (RELU) {
      f.x = fmaxf(f.x, 0.f); f.y = fmaxf(f.y, 0.f);
      f.z = fmaxf(f.z, 0.f); f.w = fmaxf(f.w, 0.f);
    }
    uint2 o;
    o.x = (unsigned)f2b(f.x) | ((unsigned)f2b(f.y) << 16);
    o.y = (unsigned)f2b(f.z) | ((unsigned)f2b(f.w) << 16);
    ((uint2*)out)[i] = o;
  }
}

// out[r, c] = bA[c] (+ bB[c]); cols is a power of two (128 or 64)
__global__ void init_bias2(float* __restrict__ out, const float* __restrict__ bA,
                           const float* __restrict__ bB, size_t n, int colMask) {
  for (size_t i = blockIdx.x * (size_t)blockDim.x + threadIdx.x; i < n;
       i += (size_t)gridDim.x * blockDim.x) {
    int c = (int)(i & (size_t)colMask);
    float v = bA[c];
    if (bB) v += bB[c];
    out[i] = v;
  }
}

// Convert W[128 x NCOLS] f32 row-major -> bf16 pre-swizzled into per-lane WMMA
// B-fragment order: out[((kt*CT + nt)*512) + L*16 + i] = bf16(W[kt*32 + 8*(L/16) + (i%8) + 16*(i/8)][nt*16 + L%16])
// so the GEMM reads each fragment as one contiguous 32-byte vector per lane.
template <int NCOLS>
__global__ void cvt_w_swizzle(const float* __restrict__ W, unsigned short* __restrict__ out) {
  constexpr int CT = NCOLS / 16;
  const int n = 128 * NCOLS;
  for (int t = blockIdx.x * blockDim.x + threadIdx.x; t < n; t += gridDim.x * blockDim.x) {
    const int i = t & 15;
    const int L = (t >> 4) & 31;
    const int rest = t >> 9;          // kt*CT + nt
    const int nt = rest % CT;
    const int kt = rest / CT;
    const int krow = kt * 32 + (L >> 4) * 8 + (i & 7) + (i >> 3) * 16;
    const int col = nt * 16 + (L & 15);
    out[t] = f2b(W[krow * NCOLS + col]);
  }
}

// ---------------- WMMA GEMM: C[M x NCOLS] = A[M x 128] @ B[128 x NCOLS], all bf16 ----------
// One wave owns a 16-row stripe and ALL column tiles: per k-step one A fragment
// (two b128 loads/lane — the ISA 16-bit A 16x32 layout is lane-contiguous in two
// 16B chunks) feeds CT back-to-back v_wmma_f32_16x16x32_bf16 ops with pre-swizzled
// B fragments (one 32B vector load/lane each). Inner loop is pure b128 + wmma.
template <int NCOLS>
__global__ __launch_bounds__(256) void gemm_wmma(const unsigned short* __restrict__ A,
                                                 const unsigned short* __restrict__ Bsw,
                                                 unsigned short* __restrict__ C, int M) {
  constexpr int K = 128;
  constexpr int CT = NCOLS / 16;
  const int wave = blockIdx.x * (blockDim.x >> 5) + (threadIdx.x >> 5);
  if (wave >= (M >> 4)) return;  // wave-uniform exit: surviving waves keep EXEC all-1s
  const int lane = threadIdx.x & 31;
  const int grp  = lane >> 4;
  const int mr   = lane & 15;
  const int m0   = wave << 4;
  const size_t rowOff = (size_t)(m0 + mr) * K;

  v8f acc[CT];
#pragma unroll
  for (int t = 0; t < CT; ++t) acc[t] = (v8f){0.f, 0.f, 0.f, 0.f, 0.f, 0.f, 0.f, 0.f};

#pragma unroll
  for (int kt = 0; kt < 4; ++kt) {
    const int k0 = kt * 32;
    const unsigned short* pa = A + rowOff + k0 + grp * 8;
    const uint4 c0 = *(const uint4*)pa;          // K = k0+grp*8 .. +7
    const uint4 c1 = *(const uint4*)(pa + 16);   // K = k0+grp*8+16 .. +23
    const v8u ar = {c0.x, c0.y, c0.z, c0.w, c1.x, c1.y, c1.z, c1.w};
    const v16bf a = __builtin_bit_cast(v16bf, ar);
#pragma unroll
    for (int nt = 0; nt < CT; ++nt) {
      const v8u braw = *(const v8u*)(Bsw + (size_t)(kt * CT + nt) * 512 + lane * 16);
      acc[nt] = __builtin_amdgcn_wmma_f32_16x16x32_bf16(
          false, a, false, __builtin_bit_cast(v16bf, braw), (short)0, acc[nt], false, false);
    }
  }

#pragma unroll
  for (int nt = 0; nt < CT; ++nt) {
#pragma unroll
    for (int r = 0; r < 8; ++r) {
      const int row = m0 + r + grp * 8;
      C[(size_t)row * NCOLS + nt * 16 + mr] = f2b(acc[nt][r]);
    }
  }
}

// ---------------- edge scatter: agg[dst] += y[src] * nsrc[src]*ndst[dst] ----------------
// One wave per edge; each lane loads W/32 contiguous bf16 (b64/b32 vector load)
// and issues f32 atomics (land in L2: dst working set <= 153MB < 192MB L2).
template <int W>
__global__ __launch_bounds__(256) void scatter_edges(const unsigned short* __restrict__ y,
                                                     const int* __restrict__ src,
                                                     const int* __restrict__ dst,
                                                     const float* __restrict__ nsrc,
                                                     const float* __restrict__ ndst,
                                                     float* __restrict__ agg, int E) {
  const int wave = blockIdx.x * (blockDim.x >> 5) + (threadIdx.x >> 5);
  if (wave >= E) return;
  const int lane = threadIdx.x & 31;
  const int s = src[wave];
  const int d = dst[wave];
  const float sc = nsrc[s] * ndst[d];
  constexpr int PER = W / 32;  // 4 (layer1) or 2 (layer2)
  const unsigned short* yr = y + (size_t)s * W + lane * PER;
  float* ar = agg + (size_t)d * W + lane * PER;
  if constexpr (PER == 4) {
    const uint2 v = *(const uint2*)yr;
    atomAddF(ar + 0, b2f((unsigned short)(v.x & 0xffffu)) * sc);
    atomAddF(ar + 1, b2f((unsigned short)(v.x >> 16)) * sc);
    atomAddF(ar + 2, b2f((unsigned short)(v.y & 0xffffu)) * sc);
    atomAddF(ar + 3, b2f((unsigned short)(v.y >> 16)) * sc);
  } else {
    const unsigned v = *(const unsigned*)yr;
    atomAddF(ar + 0, b2f((unsigned short)(v & 0xffffu)) * sc);
    atomAddF(ar + 1, b2f((unsigned short)(v >> 16)) * sc);
  }
}

// ---------------- launch ----------------

static inline unsigned gsz(size_t n, unsigned t) {
  size_t b = (n + t - 1) / t;
  if (b > 1048576u) b = 1048576u;  // grid-stride covers the rest
  return (unsigned)b;
}

extern "C" void kernel_launch(void* const* d_in, const int* in_sizes, int n_in,
                              void* d_out, int out_size, void* d_ws, size_t ws_size,
                              hipStream_t stream) {
  (void)in_sizes; (void)n_in; (void)out_size; (void)ws_size;

  const float* x_user = (const float*)d_in[0];
  const float* x_item = (const float*)d_in[1];
  const int* ra_src = (const int*)d_in[2];
  const int* ra_dst = (const int*)d_in[3];
  const int* rb_src = (const int*)d_in[4];
  const int* rb_dst = (const int*)d_in[5];
  const int* fo_src = (const int*)d_in[6];
  const int* fo_dst = (const int*)d_in[7];
  const float* W1_ra = (const float*)d_in[8];
  const float* b1_ra = (const float*)d_in[9];
  const float* W1_rb = (const float*)d_in[10];
  const float* b1_rb = (const float*)d_in[11];
  const float* W1_fo = (const float*)d_in[12];
  const float* b1_fo = (const float*)d_in[13];
  const float* W2_ra = (const float*)d_in[14];
  const float* b2_ra = (const float*)d_in[15];
  const float* W2_rb = (const float*)d_in[16];
  const float* b2_rb = (const float*)d_in[17];
  const float* W2_fo = (const float*)d_in[18];
  const float* b2_fo = (const float*)d_in[19];

  // ---- workspace carve-out ----
  size_t off = 0;
  auto take = [&](size_t bytes) -> char* {
    off = (off + 255) & ~(size_t)255;
    char* p = (char*)d_ws + off;
    off += bytes;
    return p;
  };
  float* nrm = (float*)take((size_t)(3 * NU + 2 * NI + NU) * sizeof(float));  // 800k floats
  float* n_src_ra = nrm;               // NU
  float* n_dst_ra = n_src_ra + NU;     // NI
  float* n_src_rb = n_dst_ra + NI;     // NI
  float* n_dst_rb = n_src_rb + NI;     // NU
  float* n_src_fo = n_dst_rb + NU;     // NU
  float* n_dst_fo = n_src_fo + NU;     // NU

  unsigned short* w1ra = (unsigned short*)take((size_t)DIN * DH * 2);
  unsigned short* w1rb = (unsigned short*)take((size_t)DIN * DH * 2);
  unsigned short* w1fo = (unsigned short*)take((size_t)DIN * DH * 2);
  unsigned short* w2ra = (unsigned short*)take((size_t)DH * DOUT * 2);
  unsigned short* w2rb = (unsigned short*)take((size_t)DH * DOUT * 2);
  unsigned short* w2fo = (unsigned short*)take((size_t)DH * DOUT * 2);

  unsigned short* y1_ra = (unsigned short*)take((size_t)NU * DH * 2);
  unsigned short* y1_rb = (unsigned short*)take((size_t)NI * DH * 2);
  unsigned short* y1_fo = (unsigned short*)take((size_t)NU * DH * 2);
  // layer-2 y buffers alias layer-1 (dead after layer-1 scatter), 64-wide fits
  unsigned short* y2_ra = y1_ra;
  unsigned short* y2_rb = y1_rb;
  unsigned short* y2_fo = y1_fo;

  float* h_item_f = (float*)take((size_t)NI * DH * sizeof(float));
  float* h_user_f = (float*)take((size_t)NU * DH * sizeof(float));
  unsigned short* h_item_b = (unsigned short*)take((size_t)NI * DH * 2);
  unsigned short* h_user_b = (unsigned short*)take((size_t)NU * DH * 2);

  // bf16 copies of x alias the h_item_f region: dead before h_item_f is first written
  // (offset NU*DH*2 = 25.6e6 bytes, 256B-aligned, 76.8MB <= 102.4MB region)
  unsigned short* xu_b = (unsigned short*)h_item_f;
  unsigned short* xi_b = xu_b + (size_t)NU * DH;

  float* o_user = (float*)d_out;               // NU x 64
  float* o_item = o_user + (size_t)NU * DOUT;  // NI x 64

  const size_t NRM_N = (size_t)3 * NU + 2 * NI + NU;

  // ---- degrees -> rsqrt norms ----
  fill_f32<<<gsz(NRM_N, 256), 256, 0, stream>>>(nrm, NRM_N, 0.0f);
  degree_kernel<<<gsz(EE, 256), 256, 0, stream>>>(ra_src, n_src_ra, EE);
  degree_kernel<<<gsz(EE, 256), 256, 0, stream>>>(ra_dst, n_dst_ra, EE);
  degree_kernel<<<gsz(EE, 256), 256, 0, stream>>>(rb_src, n_src_rb, EE);
  degree_kernel<<<gsz(EE, 256), 256, 0, stream>>>(rb_dst, n_dst_rb, EE);
  degree_kernel<<<gsz(EE, 256), 256, 0, stream>>>(fo_src, n_src_fo, EE);
  degree_kernel<<<gsz(EE, 256), 256, 0, stream>>>(fo_dst, n_dst_fo, EE);
  rsqrt_kernel<<<gsz(NRM_N, 256), 256, 0, stream>>>(nrm, NRM_N);

  // ---- weights -> bf16, pre-swizzled into WMMA B-fragment order ----
  cvt_w_swizzle<DH><<<64, 256, 0, stream>>>(W1_ra, w1ra);
  cvt_w_swizzle<DH><<<64, 256, 0, stream>>>(W1_rb, w1rb);
  cvt_w_swizzle<DH><<<64, 256, 0, stream>>>(W1_fo, w1fo);
  cvt_w_swizzle<DOUT><<<32, 256, 0, stream>>>(W2_ra, w2ra);
  cvt_w_swizzle<DOUT><<<32, 256, 0, stream>>>(W2_rb, w2rb);
  cvt_w_swizzle<DOUT><<<32, 256, 0, stream>>>(W2_fo, w2fo);

  // ---- x -> bf16 (vectorized, one pass) ----
  cvt_bf16x4<false><<<gsz((size_t)NU * DIN / 4, 256), 256, 0, stream>>>(x_user, xu_b, (size_t)NU * DIN / 4);
  cvt_bf16x4<false><<<gsz((size_t)NI * DIN / 4, 256), 256, 0, stream>>>(x_item, xi_b, (size_t)NI * DIN / 4);

  // ---- layer 1: y = x @ W1 (pure bf16 WMMA), GEMM-first ----
  {
    const unsigned wavesU = NU / 16;   // one wave per 16-row stripe (all col tiles)
    const unsigned wavesI = NI / 16;
    gemm_wmma<DH><<<(wavesU + 7) / 8, 256, 0, stream>>>(xu_b, w1ra, y1_ra, NU);
    gemm_wmma<DH><<<(wavesI + 7) / 8, 256, 0, stream>>>(xi_b, w1rb, y1_rb, NI);
    gemm_wmma<DH><<<(wavesU + 7) / 8, 256, 0, stream>>>(xu_b, w1fo, y1_fo, NU);
  }

  // ---- init h with biases (x bf16 copies now dead; h_item_f region reused), scatter ----
  init_bias2<<<gsz((size_t)NI * DH, 256), 256, 0, stream>>>(h_item_f, b1_ra, nullptr,
                                                            (size_t)NI * DH, DH - 1);
  init_bias2<<<gsz((size_t)NU * DH, 256), 256, 0, stream>>>(h_user_f, b1_rb, b1_fo,
                                                            (size_t)NU * DH, DH - 1);
  scatter_edges<DH><<<(EE + 7) / 8, 256, 0, stream>>>(y1_ra, ra_src, ra_dst, n_src_ra, n_dst_ra, h_item_f, EE);
  scatter_edges<DH><<<(EE + 7) / 8, 256, 0, stream>>>(y1_rb, rb_src, rb_dst, n_src_rb, n_dst_rb, h_user_f, EE);
  scatter_edges<DH><<<(EE + 7) / 8, 256, 0, stream>>>(y1_fo, fo_src, fo_dst, n_src_fo, n_dst_fo, h_user_f, EE);

  // ---- relu + bf16 (vectorized) ----
  cvt_bf16x4<true><<<gsz((size_t)NI * DH / 4, 256), 256, 0, stream>>>(h_item_f, h_item_b, (size_t)NI * DH / 4);
  cvt_bf16x4<true><<<gsz((size_t)NU * DH / 4, 256), 256, 0, stream>>>(h_user_f, h_user_b, (size_t)NU * DH / 4);

  // ---- layer 2: y2 = h @ W2 (pure bf16 WMMA) ----
  {
    const unsigned wavesU = NU / 16;
    const unsigned wavesI = NI / 16;
    gemm_wmma<DOUT><<<(wavesU + 7) / 8, 256, 0, stream>>>(h_user_b, w2ra, y2_ra, NU);
    gemm_wmma<DOUT><<<(wavesI + 7) / 8, 256, 0, stream>>>(h_item_b, w2rb, y2_rb, NI);
    gemm_wmma<DOUT><<<(wavesU + 7) / 8, 256, 0, stream>>>(h_user_b, w2fo, y2_fo, NU);
  }

  // ---- init outputs with biases, scatter into d_out ----
  init_bias2<<<gsz((size_t)NU * DOUT, 256), 256, 0, stream>>>(o_user, b2_rb, b2_fo,
                                                              (size_t)NU * DOUT, DOUT - 1);
  init_bias2<<<gsz((size_t)NI * DOUT, 256), 256, 0, stream>>>(o_item, b2_ra, nullptr,
                                                              (size_t)NI * DOUT, DOUT - 1);
  scatter_edges<DOUT><<<(EE + 7) / 8, 256, 0, stream>>>(y2_ra, ra_src, ra_dst, n_src_ra, n_dst_ra, o_item, EE);
  scatter_edges<DOUT><<<(EE + 7) / 8, 256, 0, stream>>>(y2_rb, rb_src, rb_dst, n_src_rb, n_dst_rb, o_user, EE);
  scatter_edges<DOUT><<<(EE + 7) / 8, 256, 0, stream>>>(y2_fo, fo_src, fo_dst, n_src_fo, n_dst_fo, o_user, EE);
}
